// DWRBlock_51281909514480
// MI455X (gfx1250) — compile-verified, matrix operand order checked by reading, not run on hardware
//
#include <hip/hip_runtime.h>
#include <hip/hip_bf16.h>

// ---------------------------------------------------------------------------
// MoE block (gate -> top-2 -> dense-equivalent experts -> residual+LN) for
// gfx1250. Matmuls via v_wmma_f32_16x16x32_bf16; FFN2 A-tile staging via
// global_load_async_to_lds_b128 (ASYNCcnt path).
// ---------------------------------------------------------------------------

#define D_MODEL   1024
#define D_FF      4096
#define N_EXP     8
#define T_TOKENS  8192
#define LN_EPS    1e-5f

typedef __bf16 bf16;
typedef __attribute__((ext_vector_type(16))) __bf16 v16bf;
typedef __attribute__((ext_vector_type(8)))  __bf16 v8bf;
typedef __attribute__((ext_vector_type(8)))  float  v8f;

#define BM 128
#define BN 128
#define BK 64   // two WMMA K-steps (2 x 32) per staged tile

// ---------------------------------------------------------------------------
// gfx1250 async copy helpers (inline asm: portable across ROCm/upstream
// toolchains per CDNA5 bridge doc). Each lane copies 16B global -> LDS,
// tracked on ASYNCcnt.
// ---------------------------------------------------------------------------
__device__ __forceinline__ void async_copy_b128(const void* gptr, void* lptr)
{
    unsigned lds32 = (unsigned)(uintptr_t)lptr;   // LDS flat addr = addr[31:0]
    asm volatile("global_load_async_to_lds_b128 %0, %1, off"
                 :: "v"(lds32), "v"(gptr) : "memory");
}

__device__ __forceinline__ void async_wait0()
{
    asm volatile("s_wait_asynccnt 0" ::: "memory");
}

// ---------------------------------------------------------------------------
// Gating: one wave32 per token. logits = x[t,:] @ Wg + bg ; softmax ; top-2 ;
// wts[t,e] = prob if e in top2 else 0. Accumulate f_e counts and P_e sums.
// ---------------------------------------------------------------------------
__global__ __launch_bounds__(256) void gate_kernel(
    const float* __restrict__ X, const float* __restrict__ Wg,
    const float* __restrict__ bg, float* __restrict__ wts,
    float* __restrict__ fP /* [0..7]=f counts, [8..15]=P sums */)
{
    const int lane = threadIdx.x & 31;
    const int wid  = threadIdx.x >> 5;
    const int t    = blockIdx.x * 8 + wid;

    float lg[N_EXP];
#pragma unroll
    for (int e = 0; e < N_EXP; ++e) lg[e] = 0.f;

    const float* xp = X + (size_t)t * D_MODEL;
    for (int d = lane; d < D_MODEL; d += 32) {
        float xv = xp[d];
#pragma unroll
        for (int e = 0; e < N_EXP; ++e) lg[e] += xv * Wg[d * N_EXP + e];
    }
#pragma unroll
    for (int off = 16; off > 0; off >>= 1) {
#pragma unroll
        for (int e = 0; e < N_EXP; ++e) lg[e] += __shfl_xor(lg[e], off, 32);
    }
#pragma unroll
    for (int e = 0; e < N_EXP; ++e) lg[e] += bg[e];

    float mx = lg[0];
#pragma unroll
    for (int e = 1; e < N_EXP; ++e) mx = fmaxf(mx, lg[e]);
    float p[N_EXP], s = 0.f;
#pragma unroll
    for (int e = 0; e < N_EXP; ++e) { p[e] = __expf(lg[e] - mx); s += p[e]; }
    const float inv = 1.f / s;
#pragma unroll
    for (int e = 0; e < N_EXP; ++e) p[e] *= inv;

    // top-2, ties keep lower index first (matches lax.top_k)
    int i1 = 0, i2 = -1;
    float b1v = p[0], b2v = -1.f;
#pragma unroll
    for (int e = 1; e < N_EXP; ++e) {
        if (p[e] > b1v)      { b2v = b1v; i2 = i1; b1v = p[e]; i1 = e; }
        else if (p[e] > b2v) { b2v = p[e]; i2 = e; }
    }

    if (lane == 0) {
#pragma unroll
        for (int e = 0; e < N_EXP; ++e)
            wts[(size_t)t * N_EXP + e] = (e == i1) ? b1v : (e == i2) ? b2v : 0.f;
        atomicAdd(&fP[i1], 1.f);
        atomicAdd(&fP[i2], 1.f);
#pragma unroll
        for (int e = 0; e < N_EXP; ++e) atomicAdd(&fP[8 + e], p[e]);
    }
}

__global__ void aux_kernel(const float* __restrict__ fP, float* __restrict__ out_aux)
{
    if (threadIdx.x == 0 && blockIdx.x == 0) {
        float s = 0.f;
#pragma unroll
        for (int e = 0; e < N_EXP; ++e)
            s += (fP[e] / (float)(T_TOKENS * 2)) * (fP[8 + e] / (float)T_TOKENS);
        *out_aux = (float)N_EXP * s;
    }
}

// ---------------------------------------------------------------------------
// Fragment helpers. ISA 16-bit A-matrix 16x32 layout: lanes 0-15 (rows M) hold
// K={0..7,16..23}, lanes 16-31 hold K={8..15,24..31}; each half is 8
// contiguous bf16 (16B) in row-major LDS -> two ds_load_b128 per fragment.
// B tiles are staged transposed ([N][K]) so B fragments load identically.
// ---------------------------------------------------------------------------
__device__ __forceinline__ v16bf load_frag(const bf16* rowptr, int kb)
{
    v8bf lo = *(const v8bf*)(rowptr + kb);
    v8bf hi = *(const v8bf*)(rowptr + kb + 16);
    return __builtin_shufflevector(lo, hi, 0,1,2,3,4,5,6,7,8,9,10,11,12,13,14,15);
}

// ---------------------------------------------------------------------------
// FFN1: H[t,n] = relu(X @ W1e + b1e), stored bf16. M=8192 K=1024 N=4096.
// Block = 8 waves, tile 128x128x64; wave computes 64x32 (acc v8f[4][2]).
// ---------------------------------------------------------------------------
__global__ __launch_bounds__(256) void ffn1_kernel(
    const float* __restrict__ X, const float* __restrict__ W1e,
    const float* __restrict__ b1e, bf16* __restrict__ H)
{
    __shared__ bf16 As[BM][BK];
    __shared__ bf16 Bs[BN][BK];   // transposed: Bs[n][k]

    const int tid  = threadIdx.x;
    const int lane = tid & 31;
    const int wid  = tid >> 5;
    const int half = lane >> 4;
    const int l16  = lane & 15;
    const int waveM = wid >> 2;   // 0..1 -> 64 rows each
    const int waveN = wid & 3;    // 0..3 -> 32 cols each
    const int m0 = blockIdx.y * BM;
    const int n0 = blockIdx.x * BN;

    v8f acc[4][2];
#pragma unroll
    for (int mi = 0; mi < 4; ++mi)
#pragma unroll
        for (int ni = 0; ni < 2; ++ni) acc[mi][ni] = {};

    for (int k0 = 0; k0 < D_MODEL; k0 += BK) {
        // stage A: 128x64 f32 -> bf16 (16 float4 per row)
#pragma unroll
        for (int j = 0; j < 8; ++j) {
            int idx = tid + j * 256;          // 0..2047
            int row = idx >> 4;
            int c4  = idx & 15;
            float4 v = ((const float4*)(X + (size_t)(m0 + row) * D_MODEL + k0))[c4];
            As[row][c4 * 4 + 0] = (bf16)v.x;
            As[row][c4 * 4 + 1] = (bf16)v.y;
            As[row][c4 * 4 + 2] = (bf16)v.z;
            As[row][c4 * 4 + 3] = (bf16)v.w;
        }
        // stage B transposed: 64(K) x 128(N) f32 -> Bs[n][k]
#pragma unroll
        for (int j = 0; j < 8; ++j) {
            int idx = tid + j * 256;
            int kr  = idx >> 5;
            int c4  = idx & 31;
            float4 v = ((const float4*)(W1e + (size_t)(k0 + kr) * D_FF + n0))[c4];
            Bs[c4 * 4 + 0][kr] = (bf16)v.x;
            Bs[c4 * 4 + 1][kr] = (bf16)v.y;
            Bs[c4 * 4 + 2][kr] = (bf16)v.z;
            Bs[c4 * 4 + 3][kr] = (bf16)v.w;
        }
        if (k0 + BK < D_MODEL) {   // global_prefetch_b8 for next stage
            __builtin_prefetch(X   + (size_t)(m0 + (tid >> 1)) * D_MODEL + k0 + BK, 0, 0);
            __builtin_prefetch(W1e + (size_t)(k0 + BK + (tid & 63)) * D_FF + n0, 0, 0);
        }
        __syncthreads();
#pragma unroll
        for (int kk = 0; kk < 2; ++kk) {
            const int kb = kk * 32 + half * 8;
            v16bf a[4], b[2];
#pragma unroll
            for (int mi = 0; mi < 4; ++mi)
                a[mi] = load_frag(&As[waveM * 64 + mi * 16 + l16][0], kb);
#pragma unroll
            for (int ni = 0; ni < 2; ++ni)
                b[ni] = load_frag(&Bs[waveN * 32 + ni * 16 + l16][0], kb);
#pragma unroll
            for (int mi = 0; mi < 4; ++mi)
#pragma unroll
                for (int ni = 0; ni < 2; ++ni)
                    acc[mi][ni] = __builtin_amdgcn_wmma_f32_16x16x32_bf16(
                        false, a[mi], false, b[ni], (short)0, acc[mi][ni], false, false);
        }
        __syncthreads();
    }

    // epilogue: +bias, relu, bf16 store. C layout: VGPR r -> M = r + half*8.
#pragma unroll
    for (int mi = 0; mi < 4; ++mi)
#pragma unroll
        for (int ni = 0; ni < 2; ++ni) {
            const int n = n0 + waveN * 32 + ni * 16 + l16;
            const float bias = b1e[n];
#pragma unroll
            for (int r = 0; r < 8; ++r) {
                const int m = m0 + waveM * 64 + mi * 16 + half * 8 + r;
                float v = acc[mi][ni][r] + bias;
                v = fmaxf(v, 0.f);
                H[(size_t)m * D_FF + n] = (bf16)v;
            }
        }
}

// ---------------------------------------------------------------------------
// FFN2: accOut[t,n] (+)= wts[t,e] * (H @ W2e + b2e).  M=8192 K=4096 N=1024.
// A tile (bf16, no conversion needed) staged with async global->LDS copies.
// ---------------------------------------------------------------------------
__global__ __launch_bounds__(256) void ffn2_kernel(
    const bf16* __restrict__ H, const float* __restrict__ W2e,
    const float* __restrict__ b2e, const float* __restrict__ wts,
    int expert, int isFirst, float* __restrict__ accOut)
{
    __shared__ bf16 As[BM][BK];
    __shared__ bf16 Bs[BN][BK];

    const int tid  = threadIdx.x;
    const int lane = tid & 31;
    const int wid  = tid >> 5;
    const int half = lane >> 4;
    const int l16  = lane & 15;
    const int waveM = wid >> 2;
    const int waveN = wid & 3;
    const int m0 = blockIdx.y * BM;
    const int n0 = blockIdx.x * BN;

    v8f acc[4][2];
#pragma unroll
    for (int mi = 0; mi < 4; ++mi)
#pragma unroll
        for (int ni = 0; ni < 2; ++ni) acc[mi][ni] = {};

    for (int k0 = 0; k0 < D_FF; k0 += BK) {
        // stage A: bf16 straight copy via async global->LDS (16B per lane,
        // ASYNCcnt); 128 rows x 64 bf16 = 1024 x 16B chunks, 4 per thread.
#pragma unroll
        for (int j = 0; j < 4; ++j) {
            int idx = tid + j * 256;          // 0..1023
            int row = idx >> 3;
            int c   = idx & 7;
            async_copy_b128(H + (size_t)(m0 + row) * D_FF + k0 + c * 8,
                            &As[row][c * 8]);
        }
        // stage B transposed: 64(K) x 128(N) f32 -> bf16
#pragma unroll
        for (int j = 0; j < 8; ++j) {
            int idx = tid + j * 256;
            int kr  = idx >> 5;
            int c4  = idx & 31;
            float4 v = ((const float4*)(W2e + (size_t)(k0 + kr) * D_MODEL + n0))[c4];
            Bs[c4 * 4 + 0][kr] = (bf16)v.x;
            Bs[c4 * 4 + 1][kr] = (bf16)v.y;
            Bs[c4 * 4 + 2][kr] = (bf16)v.z;
            Bs[c4 * 4 + 3][kr] = (bf16)v.w;
        }
        if (k0 + BK < D_FF) {
            __builtin_prefetch(H   + (size_t)(m0 + (tid >> 1)) * D_FF + k0 + BK, 0, 0);
            __builtin_prefetch(W2e + (size_t)(k0 + BK + (tid & 63)) * D_MODEL + n0, 0, 0);
        }
        async_wait0();          // own-wave async copies landed in LDS
        __syncthreads();        // all waves' staging visible
#pragma unroll
        for (int kk = 0; kk < 2; ++kk) {
            const int kb = kk * 32 + half * 8;
            v16bf a[4], b[2];
#pragma unroll
            for (int mi = 0; mi < 4; ++mi)
                a[mi] = load_frag(&As[waveM * 64 + mi * 16 + l16][0], kb);
#pragma unroll
            for (int ni = 0; ni < 2; ++ni)
                b[ni] = load_frag(&Bs[waveN * 32 + ni * 16 + l16][0], kb);
#pragma unroll
            for (int mi = 0; mi < 4; ++mi)
#pragma unroll
                for (int ni = 0; ni < 2; ++ni)
                    acc[mi][ni] = __builtin_amdgcn_wmma_f32_16x16x32_bf16(
                        false, a[mi], false, b[ni], (short)0, acc[mi][ni], false, false);
        }
        __syncthreads();
    }

#pragma unroll
    for (int mi = 0; mi < 4; ++mi)
#pragma unroll
        for (int ni = 0; ni < 2; ++ni) {
            const int n = n0 + waveN * 32 + ni * 16 + l16;
            const float bias = b2e[n];
#pragma unroll
            for (int r = 0; r < 8; ++r) {
                const int m = m0 + waveM * 64 + mi * 16 + half * 8 + r;
                const float wv = wts[(size_t)m * N_EXP + expert];
                const size_t o = (size_t)m * D_MODEL + n;
                const float y = acc[mi][ni][r] + bias;
                const float prev = isFirst ? 0.f : accOut[o];
                accOut[o] = prev + wv * y;
            }
        }
}

// ---------------------------------------------------------------------------
// Residual + post-LayerNorm: one block per token, 4 elems/thread.
// ---------------------------------------------------------------------------
__global__ __launch_bounds__(256) void ln_kernel(
    const float* __restrict__ X, const float* __restrict__ acc,
    const float* __restrict__ gamma, const float* __restrict__ beta,
    float* __restrict__ Z)
{
    __shared__ float red[256];
    const int t = blockIdx.x;
    const float* xp = X   + (size_t)t * D_MODEL;
    const float* ap = acc + (size_t)t * D_MODEL;

    float z[4];
#pragma unroll
    for (int j = 0; j < 4; ++j) {
        int d = threadIdx.x + j * 256;
        z[j] = xp[d] + ap[d];
    }
    float s = z[0] + z[1] + z[2] + z[3];
    red[threadIdx.x] = s; __syncthreads();
    for (int off = 128; off > 0; off >>= 1) {
        if (threadIdx.x < off) red[threadIdx.x] += red[threadIdx.x + off];
        __syncthreads();
    }
    const float mu = red[0] * (1.f / D_MODEL);
    __syncthreads();

    float q = 0.f;
#pragma unroll
    for (int j = 0; j < 4; ++j) { float d = z[j] - mu; q += d * d; }
    red[threadIdx.x] = q; __syncthreads();
    for (int off = 128; off > 0; off >>= 1) {
        if (threadIdx.x < off) red[threadIdx.x] += red[threadIdx.x + off];
        __syncthreads();
    }
    const float rstd = rsqrtf(red[0] * (1.f / D_MODEL) + LN_EPS);
#pragma unroll
    for (int j = 0; j < 4; ++j) {
        int d = threadIdx.x + j * 256;
        Z[(size_t)t * D_MODEL + d] = (z[j] - mu) * rstd * gamma[d] + beta[d];
    }
}

// ---------------------------------------------------------------------------
extern "C" void kernel_launch(void* const* d_in, const int* in_sizes, int n_in,
                              void* d_out, int out_size, void* d_ws, size_t ws_size,
                              hipStream_t stream)
{
    const float* x     = (const float*)d_in[0];
    const float* Wg    = (const float*)d_in[1];
    const float* bg    = (const float*)d_in[2];
    const float* W1    = (const float*)d_in[3];
    const float* b1    = (const float*)d_in[4];
    const float* W2    = (const float*)d_in[5];
    const float* b2    = (const float*)d_in[6];
    const float* gamma = (const float*)d_in[7];
    const float* beta  = (const float*)d_in[8];

    float* zout = (float*)d_out;
    float* aux  = zout + (size_t)T_TOKENS * D_MODEL;   // tuple: (z, aux_loss)

    char* ws = (char*)d_ws;
    float* fP     = (float*)ws;                                       // 16 f32
    float* wts    = (float*)(ws + 1024);                              // T*8 f32
    bf16*  Hbuf   = (bf16*)(ws + (1 << 20));                          // 64 MiB
    float* accOut = (float*)(ws + (1 << 20) + (size_t)T_TOKENS * D_FF * sizeof(bf16)); // 32 MiB

    hipMemsetAsync(fP, 0, 16 * sizeof(float), stream);
    gate_kernel<<<T_TOKENS / 8, 256, 0, stream>>>(x, Wg, bg, wts, fP);
    aux_kernel<<<1, 32, 0, stream>>>(fP, aux);

    for (int e = 0; e < N_EXP; ++e) {
        const float* W1e = W1 + (size_t)e * D_MODEL * D_FF;
        const float* b1e = b1 + (size_t)e * D_FF;
        const float* W2e = W2 + (size_t)e * D_FF * D_MODEL;
        const float* b2e = b2 + (size_t)e * D_MODEL;

        dim3 g1(D_FF / BN, T_TOKENS / BM);       // 32 x 64
        ffn1_kernel<<<g1, 256, 0, stream>>>(x, W1e, b1e, Hbuf);

        dim3 g2(D_MODEL / BN, T_TOKENS / BM);    // 8 x 64
        ffn2_kernel<<<g2, 256, 0, stream>>>(Hbuf, W2e, b2e, wts, e, (e == 0) ? 1 : 0, accOut);
    }

    ln_kernel<<<T_TOKENS, 256, 0, stream>>>(x, accOut, gamma, beta, zout);
}
